// RNNCell_52201032515999
// MI455X (gfx1250) — matrix operand order repaired and checked by reading, not compiled, not run on hardware
//
#include <hip/hip_runtime.h>
#include <math.h>

typedef __attribute__((ext_vector_type(8)))  float    v8f;
typedef __attribute__((ext_vector_type(8)))  _Float16 v8h;
typedef __attribute__((ext_vector_type(16))) _Float16 v16h;

#define SEQ   512
#define BATCH 64
#define NI    1024
#define NH    1024
#define M1    (BATCH * SEQ)   // 32768 rows of the xproj GEMM
#define NBLK_RNN 16           // workgroups in the persistent recurrence kernel

static __device__ __forceinline__ v16h cat16(v8h lo, v8h hi) {
  return __builtin_shufflevector(lo, hi, 0,1,2,3,4,5,6,7,8,9,10,11,12,13,14,15);
}

// ---------------------------------------------------------------------------
// Kernel A: convert W -> f16 (split into Wx, Wh, K-contiguous rows), h0 -> f16,
// and reset the grid-barrier counter (deterministic per launch).
// ---------------------------------------------------------------------------
__global__ __launch_bounds__(256) void prep_kernel(
    const float* __restrict__ W, const float* __restrict__ h0,
    _Float16* __restrict__ Wxb, _Float16* __restrict__ Whb,
    _Float16* __restrict__ Hb, unsigned* __restrict__ cnt) {
  const unsigned idx = blockIdx.x * 256u + threadIdx.x;
  if (idx < (unsigned)(NH * NI)) {
    const unsigned n = idx >> 10, k = idx & 1023u;
    const float* row = W + (size_t)n * (NI + NH);
    Wxb[idx] = (_Float16)row[k];
    Whb[idx] = (_Float16)row[NI + k];
  }
  if (idx < (unsigned)(BATCH * NH)) Hb[idx] = (_Float16)h0[idx];
  if (idx == 0) *cnt = 0u;
}

// ---------------------------------------------------------------------------
// Kernel B: xproj[m][n] = sum_k x[m][k] * Wx[n][k] + b[n]   (written into y)
// 128x128 tile per 256-thread block; 8 waves in a 4(M) x 2(N) grid;
// each wave: 2 (M) x 4 (N) accumulators of v_wmma_f32_16x16x32_f16.
// A (x, fp32) is converted to f16 while staging into padded LDS.
// B (Wx, f16, 2 MB) is read directly (L2-resident, heavy cross-block reuse).
// ---------------------------------------------------------------------------
#define LDA 40  // LDS row stride in halfs (80 B): conflict-free b128 reads

__global__ __launch_bounds__(256) void xproj_kernel(
    const float* __restrict__ x, const _Float16* __restrict__ Wxb,
    const float* __restrict__ bias, float* __restrict__ y) {
  __shared__ _Float16 As[128 * LDA];

  const int tid  = threadIdx.x;
  const int lane = tid & 31;
  const int w    = tid >> 5;
  const int lr   = lane & 15;   // row/col within fragment
  const int kh   = lane >> 4;   // K-half select
  const int wm   = w & 3;       // wave M position (4)
  const int wn   = w >> 2;      // wave N position (2)
  const int m_base = blockIdx.x * 128;
  const int n_base = blockIdx.y * 128;

  v8f acc[2][4];
  const v8f vzero = {0.f, 0.f, 0.f, 0.f, 0.f, 0.f, 0.f, 0.f};
#pragma unroll
  for (int i = 0; i < 2; ++i)
#pragma unroll
    for (int j = 0; j < 4; ++j) acc[i][j] = vzero;

  for (int kc = 0; kc < NI; kc += 32) {
    __syncthreads();
    {
      // Stage 128 rows x 32 k of x, converting fp32 -> f16.
      const int r  = tid >> 1;
      const int ch = (tid & 1) << 4;  // 0 or 16 halfs
      const float* gp = x + (size_t)(m_base + r) * NI + kc + ch;
      const v8f f0 = *(const v8f*)gp;
      const v8f f1 = *(const v8f*)(gp + 8);
      *(v8h*)&As[r * LDA + ch]     = __builtin_convertvector(f0, v8h);
      *(v8h*)&As[r * LDA + ch + 8] = __builtin_convertvector(f1, v8h);
    }
    __syncthreads();

    v16h afr[2];
#pragma unroll
    for (int mf = 0; mf < 2; ++mf) {
      // A layout: lanes<16 hold K {0..15}; lanes>=16 hold K {16..31}
      const _Float16* ap = &As[(wm * 32 + mf * 16 + lr) * LDA + kh * 16];
      afr[mf] = cat16(*(const v8h*)ap, *(const v8h*)(ap + 8));
    }
#pragma unroll
    for (int nf = 0; nf < 4; ++nf) {
      // B layout: lane holds one column (= one Wx row), 16 contiguous K halfs
      const _Float16* bp =
          Wxb + (size_t)(n_base + wn * 64 + nf * 16 + lr) * NI + kc + kh * 16;
      const v16h bfr = cat16(*(const v8h*)bp, *(const v8h*)(bp + 8));
#pragma unroll
      for (int mf = 0; mf < 2; ++mf)
        acc[mf][nf] = __builtin_amdgcn_wmma_f32_16x16x32_f16(
            false, afr[mf], false, bfr, (short)0, acc[mf][nf], false, false);
    }
  }

  // Epilogue: C/D layout -> lane writes column (lane&15), rows (lane>>4)*8+r
#pragma unroll
  for (int nf = 0; nf < 4; ++nf) {
    const int n  = n_base + wn * 64 + nf * 16 + lr;
    const float bn = bias[n];
#pragma unroll
    for (int mf = 0; mf < 2; ++mf) {
      const int mb = m_base + wm * 32 + mf * 16 + kh * 8;
#pragma unroll
      for (int r = 0; r < 8; ++r)
        y[(size_t)(mb + r) * NH + n] = acc[mf][nf][r] + bn;
    }
  }
}

// ---------------------------------------------------------------------------
// Kernel C: persistent recurrence. 16 workgroups; block j owns columns
// [64j, 64j+64). Per step: Hnew = tanh(xp_t + H @ Wh^T) for its columns,
// write f16 ping-pong H state + fp32 y, then device-scope barrier.
// ---------------------------------------------------------------------------
__global__ __launch_bounds__(256) void rnn_kernel(
    const _Float16* __restrict__ Whb, _Float16* __restrict__ Hb,
    float* __restrict__ y, float* __restrict__ hlast,
    unsigned* __restrict__ cnt) {
  const int tid  = threadIdx.x;
  const int lane = tid & 31;
  const int w    = tid >> 5;
  const int lr   = lane & 15;
  const int kh   = lane >> 4;
  const int wm   = w & 3;    // 4 M-waves x 16 rows = 64 batch rows
  const int wn   = w >> 2;   // 2 N-waves x 32 cols = 64 cols
  const int n0   = blockIdx.x * 64;
  const int arow = wm * 16 + lr;
  const v8f vzero = {0.f, 0.f, 0.f, 0.f, 0.f, 0.f, 0.f, 0.f};

  for (int t = 0; t < SEQ; ++t) {
    const _Float16* Hcur = Hb + (size_t)(t & 1) * (BATCH * NH);
    _Float16*       Hnxt = Hb + (size_t)((t + 1) & 1) * (BATCH * NH);

    v8f acc[2];
    acc[0] = vzero;
    acc[1] = vzero;

#pragma unroll 4
    for (int kc = 0; kc < NH; kc += 32) {
      const _Float16* ap = Hcur + (size_t)arow * NH + kc + kh * 16;
      const v16h afr = cat16(*(const v8h*)ap, *(const v8h*)(ap + 8));
#pragma unroll
      for (int nf = 0; nf < 2; ++nf) {
        const _Float16* bp =
            Whb + (size_t)(n0 + wn * 32 + nf * 16 + lr) * NH + kc + kh * 16;
        const v16h bfr = cat16(*(const v8h*)bp, *(const v8h*)(bp + 8));
        acc[nf] = __builtin_amdgcn_wmma_f32_16x16x32_f16(
            false, afr, false, bfr, (short)0, acc[nf], false, false);
      }
    }

#pragma unroll
    for (int nf = 0; nf < 2; ++nf) {
      const int n = n0 + wn * 32 + nf * 16 + lr;
#pragma unroll
      for (int r = 0; r < 8; ++r) {
        const int m = wm * 16 + kh * 8 + r;                 // batch index
        const size_t yi = ((size_t)m * SEQ + t) * NH + n;   // y[b][t][n]
        const float v = tanhf(acc[nf][r] + y[yi]);          // xp already has b
        y[yi] = v;
        Hnxt[(size_t)m * NH + n] = (_Float16)v;
        if (t == SEQ - 1) hlast[(size_t)m * NH + n] = v;
      }
    }

    // Device-scope split barrier: release our H-tile stores, wait for all 16.
    __syncthreads();
    if (tid == 0) {
      __hip_atomic_fetch_add(cnt, 1u, __ATOMIC_RELEASE, __HIP_MEMORY_SCOPE_AGENT);
      const unsigned target = (unsigned)NBLK_RNN * (unsigned)(t + 1);
      while (__hip_atomic_load(cnt, __ATOMIC_ACQUIRE, __HIP_MEMORY_SCOPE_AGENT) < target) {
        __builtin_amdgcn_s_sleep(1);
      }
    }
    __syncthreads();
  }
}

// ---------------------------------------------------------------------------
extern "C" void kernel_launch(void* const* d_in, const int* in_sizes, int n_in,
                              void* d_out, int out_size, void* d_ws, size_t ws_size,
                              hipStream_t stream) {
  (void)in_sizes; (void)n_in; (void)out_size; (void)ws_size;
  const float* x  = (const float*)d_in[0];   // (64,512,1024)
  const float* h0 = (const float*)d_in[1];   // (64,1024)
  const float* W  = (const float*)d_in[2];   // (1024,2048)
  const float* b  = (const float*)d_in[3];   // (1024,)

  float* y     = (float*)d_out;              // (64,512,1024)
  float* hlast = y + (size_t)M1 * NH;        // (64,1024)

  _Float16* Wxb = (_Float16*)d_ws;                       // 2 MB
  _Float16* Whb = Wxb + (size_t)NH * NI;                 // 2 MB
  _Float16* Hb  = Whb + (size_t)NH * NI;                 // 2 x 128 KB ping-pong
  unsigned* cnt = (unsigned*)(Hb + 2 * (size_t)BATCH * NH);

  prep_kernel<<<(NH * NI) / 256, 256, 0, stream>>>(W, h0, Wxb, Whb, Hb, cnt);
  xproj_kernel<<<dim3(M1 / 128, NH / 128), 256, 0, stream>>>(x, Wxb, b, y);
  rnn_kernel<<<NBLK_RNN, 256, 0, stream>>>(Whb, Hb, y, hlast, cnt);
}
